// SwinWindowCrossBand_8564164789082
// MI455X (gfx1250) — matrix operand order repaired and checked by reading, not compiled, not run on hardware
//
#include <hip/hip_runtime.h>
#include <hip/hip_bf16.h>
#include <cstdint>

// ---------------------------------------------------------------------------
// SwinWindowCrossBand on gfx1250 (MI455X): f16 WMMA for projections, QK^T,
// PV, and the MLP GEMMs; async global->LDS copies + prefetch on the GEMM path.
// ---------------------------------------------------------------------------

typedef __attribute__((ext_vector_type(16))) _Float16 v16h;
typedef __attribute__((ext_vector_type(8)))  _Float16 v8h;
typedef __attribute__((ext_vector_type(8)))  float    v8f;

static constexpr int   CHN   = 96;
static constexpr int   HWDIM = 128;
static constexpr int   HWPIX = HWDIM * HWDIM;           // 16384
static constexpr int   BATCH = 2;
static constexpr int   NWIN  = 31;                      // windows per spatial dim
static constexpr float SCL   = 0.20412414523193154f;    // (96/4)^-0.5
static constexpr float LNEPS = 1e-5f;

// 32-bit LDS byte offset of a shared-memory pointer (for async-to-LDS ops)
typedef __attribute__((address_space(3))) void lds_void;
__device__ __forceinline__ unsigned to_lds(const void* p) {
  return (unsigned)(uintptr_t)(lds_void*)p;
}

// ------------------------- WMMA fragment helpers ---------------------------
// A 16x32 f16 (ISA 7.12.2): lane<16 row M=lane, halves 0-7 -> K 0..7,
//   halves 8-15 -> K 16..23 ; lane>=16 same row, K 8..15 / K 24..31.
// B 32x16 f16: lane<16 col N=lane holds K 0..7 / K 16..23,
//   lane>=16 col N=lane-16 holds K 8..15 / K 24..31 (same as A).

__device__ __forceinline__ v16h cat8(v8h lo, v8h hi) {
  return __builtin_shufflevector(lo, hi, 0, 1, 2, 3, 4, 5, 6, 7,
                                 8, 9, 10, 11, 12, 13, 14, 15);
}

// A fragment from row-major source s[r*ld + k]; ld, k0 multiples of 8.
__device__ __forceinline__ v16h frag_a(const _Float16* s, int ld, int r0, int k0, int lane) {
  const _Float16* row = s + (size_t)(r0 + (lane & 15)) * ld + k0;
  int klo = (lane & 16) ? 8 : 0;
  int khi = (lane & 16) ? 24 : 16;
  v8h lo = *(const v8h*)(row + klo);
  v8h hi = *(const v8h*)(row + khi);
  return cat8(lo, hi);
}

// B fragment where element (k, c) lives at s[c*ld + k] (K-contiguous source).
__device__ __forceinline__ v16h frag_bt(const _Float16* s, int ld, int k0, int c0, int lane) {
  const _Float16* col =
      s + (size_t)(c0 + (lane & 15)) * ld + k0 + ((lane & 16) ? 8 : 0);
  v8h lo = *(const v8h*)(col);
  v8h hi = *(const v8h*)(col + 16);
  return cat8(lo, hi);
}

__device__ __forceinline__ void store_c_f32(float* s, int ld, int r0, int c0, v8f c, int lane) {
  int n  = c0 + (lane & 15);
  int m0 = r0 + ((lane & 16) ? 8 : 0);
#pragma unroll
  for (int i = 0; i < 8; ++i) s[(size_t)(m0 + i) * ld + n] = c[i];
}

__device__ __forceinline__ v8f wmma_f16(v16h a, v16h b, v8f c) {
  return __builtin_amdgcn_wmma_f32_16x16x32_f16(false, a, false, b, (short)0, c, false, false);
}

// #windows (stride 4, size 8, 31 windows) covering shifted coord x in [0,128)
__device__ __forceinline__ int cover(int x) {
  int imax = x >> 2; if (imax > 30) imax = 30;
  int imin = (x < 8) ? 0 : ((x - 4) >> 2);
  return imax - imin + 1;
}

// ------------------------------ small kernels ------------------------------

__global__ void copy_f32_kernel(float* dst, const float* src, size_t n) {
  size_t i = (size_t)blockIdx.x * blockDim.x + threadIdx.x;
  if (i < n) dst[i] = src[i];
}
__global__ void zero_f32_kernel(float* dst, size_t n) {
  size_t i = (size_t)blockIdx.x * blockDim.x + threadIdx.x;
  if (i < n) dst[i] = 0.f;
}
__global__ void cvt_f16_kernel(_Float16* dst, const float* src, size_t n) {
  size_t i = (size_t)blockIdx.x * blockDim.x + threadIdx.x;
  if (i < n) dst[i] = (_Float16)src[i];
}
__global__ void zero_f16_kernel(_Float16* dst, size_t n) {
  size_t i = (size_t)blockIdx.x * blockDim.x + threadIdx.x;
  if (i < n) dst[i] = (_Float16)0.f;
}

// LayerNorm over channel dim, (B,C,H,W) f32 -> f16, one thread per pixel
__global__ void ln_kernel(const float* __restrict__ x, const float* __restrict__ g,
                          const float* __restrict__ bt, _Float16* __restrict__ y) {
  int p = blockIdx.x * blockDim.x + threadIdx.x;
  if (p >= BATCH * HWPIX) return;
  int b = p >> 14, pp = p & (HWPIX - 1);
  size_t base = (size_t)b * CHN * HWPIX + pp;
  float mu = 0.f;
  for (int c = 0; c < CHN; ++c) mu += x[base + (size_t)c * HWPIX];
  mu *= (1.f / CHN);
  float var = 0.f;
  for (int c = 0; c < CHN; ++c) { float d = x[base + (size_t)c * HWPIX] - mu; var += d * d; }
  var *= (1.f / CHN);
  float inv = rsqrtf(var + LNEPS);
  for (int c = 0; c < CHN; ++c)
    y[base + (size_t)c * HWPIX] =
        (_Float16)((x[base + (size_t)c * HWPIX] - mu) * inv * g[c] + bt[c]);
}

// per-(b,c) spatial mean of lnh -> pool[b*96+c]
__global__ void pool_kernel(const _Float16* __restrict__ lnh, float* __restrict__ pool) {
  __shared__ float red[256];
  int bc = blockIdx.x;
  size_t base = (size_t)bc * HWPIX;
  float s = 0.f;
  for (int i = threadIdx.x; i < HWPIX; i += 256) s += (float)lnh[base + i];
  red[threadIdx.x] = s;
  __syncthreads();
  for (int st = 128; st > 0; st >>= 1) {
    if ((int)threadIdx.x < st) red[threadIdx.x] += red[threadIdx.x + st];
    __syncthreads();
  }
  if (threadIdx.x == 0) pool[bc] = red[0] * (1.f / HWPIX);
}

// SE: z = relu(pool @ w1^T) ; s = sigmoid(z @ w2^T), per group of 32 channels
__global__ void se_kernel(const float* __restrict__ pool, const float* __restrict__ w1,
                          const float* __restrict__ w2, float* __restrict__ sout) {
  int t = threadIdx.x;
  if (t >= BATCH * CHN) return;
  int b = t / CHN, c = t % CHN, g = c / 32, cl = c % 32;
  const float* pg = pool + b * CHN + g * 32;
  float acc = 0.f;
  for (int j = 0; j < 2; ++j) {
    float z = 0.f;
    for (int i = 0; i < 32; ++i) z += pg[i] * w1[(g * 2 + j) * 32 + i];
    z = fmaxf(z, 0.f);
    acc += z * w2[(g * 32 + cl) * 2 + j];
  }
  sout[t] = 1.f / (1.f + __expf(-acc));
}

// ------------------------- windowed cross attention ------------------------
// LDS layouts (all fragment reads contiguous):
//   X  (both bands): [64 tokens][96 ch]                    (B source, K=ch)
//   q,k: [4 heads][64 tokens][32 (d pad 24->32)]           (A for QK^T; B via T)
//   v  : [4 heads][32 (d pad)][64 tokens]                  (A for PV)
//   S  : f32 [64][64] scores;  Sh: f16 [64][64] probs      (B via T for PV)

__device__ __forceinline__ void attn_dir(
    _Float16* Xbase, int xq_off, int xkv_off,
    const _Float16* __restrict__ wf, int woff,
    _Float16* qkv, float* S, _Float16* Sh, float* __restrict__ fold,
    int b, int h0, int w0, int shift, int tid) {
  int wave = tid >> 5, lane = tid & 31;

  // zero padded q/k/v (3 * 8192 halfs) with b128 stores
  v8h zz = {};
  for (int i = tid; i < 3 * 1024; i += 256) ((v8h*)qkv)[i] = zz;
  __syncthreads();

  // projections: 3 x (96x96)*(96x64), 24 C-tiles each, K loop of 3
  for (int job = wave; job < 72; job += 8) {
    int p = job / 24, t = job % 24;
    int mt = t >> 2, nt = t & 3;
    const _Float16* W = wf + woff + p * 9216;
    const _Float16* X = Xbase + ((p == 0) ? xq_off : xkv_off);
    v8f acc = {};
#pragma unroll
    for (int ks = 0; ks < 96; ks += 32) {
      v16h a  = frag_a(W, 96, mt * 16, ks, lane);
      v16h bf = frag_bt(X, 96, ks, nt * 16, lane);
      acc = wmma_f16(a, bf, acc);
    }
    int n   = nt * 16 + (lane & 15);                 // token
    int m0  = mt * 16 + ((lane & 16) ? 8 : 0);       // first of 8 channels
    int hh  = m0 / 24, rr0 = m0 - hh * 24;           // 8-run never crosses head
    if (p < 2) {                                     // q,k transposed [tok][d]
      v8h st;
#pragma unroll
      for (int i = 0; i < 8; ++i) st[i] = (_Float16)acc[i];
      *(v8h*)(qkv + p * 8192 + hh * 2048 + n * 32 + rr0) = st;
    } else {                                         // v as [d][tok]
      _Float16* dst = qkv + 2 * 8192 + hh * 2048;
#pragma unroll
      for (int i = 0; i < 8; ++i) dst[(rr0 + i) * 64 + n] = (_Float16)acc[i];
    }
  }
  __syncthreads();

  for (int h = 0; h < 4; ++h) {
    const _Float16* qt = qkv + h * 2048;             // [64][32]
    const _Float16* kt = qkv + 8192 + h * 2048;      // [64][32]
    const _Float16* vv = qkv + 16384 + h * 2048;     // [32][64]

    // S[n][m] = sum_d q[n][d] * k[m][d] ; 16 tiles over 8 waves
    for (int t = wave * 2; t < wave * 2 + 2; ++t) {
      int mt = t >> 2, nt = t & 3;
      v16h a  = frag_a(qt, 32, mt * 16, 0, lane);
      v16h bf = frag_bt(kt, 32, 0, nt * 16, lane);
      v8f acc = {};
      acc = wmma_f16(a, bf, acc);
      store_c_f32(S, 64, mt * 16, nt * 16, acc, lane);
    }
    __syncthreads();

    // softmax over m (with SCALE), one row per thread 0..63
    if (tid < 64) {
      float* row = S + tid * 64;
      float mx = -3.0e38f;
      for (int m = 0; m < 64; ++m) mx = fmaxf(mx, row[m]);
      float sm = 0.f;
      for (int m = 0; m < 64; ++m) sm += __expf((row[m] - mx) * SCL);
      float inv = 1.f / sm;
      for (int m = 0; m < 64; ++m)
        Sh[tid * 64 + m] = (_Float16)(__expf((row[m] - mx) * SCL) * inv);
    }
    __syncthreads();

    // O[d][n] = sum_m v[d][m] * Sh[n][m] ; 8 tiles, one per wave, K = 64
    {
      int mt = wave >> 2, nt = wave & 3;
      v8f acc = {};
#pragma unroll
      for (int ks = 0; ks < 64; ks += 32) {
        v16h a  = frag_a(vv, 64, mt * 16, ks, lane);
        v16h bf = frag_bt(Sh, 64, ks, nt * 16, lane);
        acc = wmma_f16(a, bf, acc);
      }
      int n  = nt * 16 + (lane & 15);
      int m0 = mt * 16 + ((lane & 16) ? 8 : 0);
      int gh = (h0 + (n >> 3) + shift) & (HWDIM - 1);
      int gw = (w0 + (n & 7) + shift) & (HWDIM - 1);
#pragma unroll
      for (int i = 0; i < 8; ++i) {
        int d = m0 + i;
        if (d < 24) {
          int ch = h * 24 + d;
          atomicAdd(fold + (((size_t)b * CHN + ch) * HWDIM + gh) * HWDIM + gw, acc[i]);
        }
      }
    }
    __syncthreads();
  }
}

__global__ __launch_bounds__(256)
void attn_kernel(const _Float16* __restrict__ lnl, const _Float16* __restrict__ lnh,
                 const float* __restrict__ ses, const _Float16* __restrict__ wf,
                 float* __restrict__ foldL, float* __restrict__ foldH, int shift) {
  extern __shared__ char smem[];
  _Float16* Xb  = (_Float16*)smem;          // [2][64][96]  (low, high)
  _Float16* qkv = Xb + 2 * 64 * CHN;        // 3*8192
  float*    S   = (float*)(qkv + 3 * 8192); // 64*64 f32
  _Float16* Sh  = (_Float16*)(S + 64 * 64); // 64*64 f16

  int wid = blockIdx.x;
  int b = wid / (NWIN * NWIN), rem = wid % (NWIN * NWIN);
  int wh = rem / NWIN, ww = rem % NWIN;
  int tid = threadIdx.x;

  // L2-resident weights: hint them in while we load the window
  __builtin_prefetch(wf + (size_t)tid * 216, 0, 3);   // 6*9216 halfs / 256 thr

  // load window token-major (shifted coords -> true coords), SE-scale high
  for (int i = tid; i < CHN * 64; i += 256) {
    int c = i >> 6, t = i & 63;
    int gh = (wh * 4 + (t >> 3) + shift) & (HWDIM - 1);
    int gw = (ww * 4 + (t & 7) + shift) & (HWDIM - 1);
    size_t idx = (((size_t)b * CHN + c) * HWDIM + gh) * HWDIM + gw;
    Xb[t * CHN + c]            = lnl[idx];
    Xb[6144 + t * CHN + c]     = (_Float16)((float)lnh[idx] * ses[b * CHN + c]);
  }
  __syncthreads();

  // low queries attend high -> foldL
  attn_dir(Xb, 0, 6144, wf, 0, qkv, S, Sh, foldL, b, wh * 4, ww * 4, shift, tid);
  __syncthreads();
  // high queries attend low -> foldH
  attn_dir(Xb, 6144, 0, wf, 3 * 9216, qkv, S, Sh, foldH, b, wh * 4, ww * 4, shift, tid);
}

// acc += ln_residual + wproj @ (fold / cnt)   (96 threads, one pixel per block)
__global__ void proj_res_kernel(const float* __restrict__ foldL, const float* __restrict__ foldH,
                                const float* __restrict__ wpL, const float* __restrict__ wpH,
                                const _Float16* __restrict__ lnl, const _Float16* __restrict__ lnh,
                                float* __restrict__ accL, float* __restrict__ accH, int shift) {
  __shared__ float fv[CHN];
  int band = blockIdx.y;
  const float* fold = band ? foldH : foldL;
  const float* wp   = band ? wpH : wpL;
  const _Float16* ln = band ? lnh : lnl;
  float* acc = band ? accH : accL;

  int pix = blockIdx.x;
  int b = pix >> 14, pp = pix & (HWPIX - 1);
  int hh = pp >> 7, wwp = pp & (HWDIM - 1);
  int hs  = (hh - shift) & (HWDIM - 1);
  int wsd = (wwp - shift) & (HWDIM - 1);
  float inv_cnt = 1.f / (float)(cover(hs) * cover(wsd));

  int o = threadIdx.x;
  size_t base = (size_t)b * CHN * HWPIX + pp;
  fv[o] = fold[base + (size_t)o * HWPIX] * inv_cnt;
  __syncthreads();
  float y = 0.f;
  const float* wr = wp + o * CHN;
  for (int c = 0; c < CHN; ++c) y += wr[c] * fv[c];
  size_t di = base + (size_t)o * HWPIX;
  acc[di] += (float)ln[di] + y;
}

// (b,c,p) f32 -> X[c][b*HWPIX+p] f16 for the MLP GEMMs
__global__ void cvtX_kernel(const float* __restrict__ acc, _Float16* __restrict__ X) {
  size_t i = (size_t)blockIdx.x * blockDim.x + threadIdx.x;
  if (i >= (size_t)BATCH * CHN * HWPIX) return;
  int b = (int)(i / ((size_t)CHN * HWPIX));
  int c = (int)((i / HWPIX) % CHN);
  int p = (int)(i % HWPIX);
  X[(size_t)c * (BATCH * HWPIX) + (size_t)b * HWPIX + p] = (_Float16)acc[i];
}

// Tiled WMMA GEMM: C(MxN) = A(MpadxK) * B(KxN), 64x64 C tile per block.
// A tile is copied global->LDS with async-to-LDS b128 (ASYNCcnt-tracked);
// B tile is transposed through registers so fragments are K-contiguous.
// mode 0: out16 = f16(gelu_exact(C + bias))
// mode 1: out32[(b,m,p)] = resid + C + bias   (N = B*HWPIX)
__global__ __launch_bounds__(256)
void gemm_kernel(const _Float16* __restrict__ A, const _Float16* __restrict__ B,
                 const float* __restrict__ bias, int M, int N, int K, int mode,
                 _Float16* __restrict__ out16, const float* __restrict__ resid,
                 float* __restrict__ out32) {
  __shared__ _Float16 As[64 * 32];    // [m][k]
  __shared__ _Float16 Bst[64 * 32];   // [n][k]  (transposed tile)
  int tid = threadIdx.x, wave = tid >> 5, lane = tid & 31;
  int m0 = blockIdx.x * 64, n0 = blockIdx.y * 64;
  int mt = wave >> 1, ntb = (wave & 1) * 2;
  int ra = tid >> 2, ka = (tid & 3) * 8;   // A-tile copy coords (this thread)
  int rb = tid >> 3, cb = (tid & 7) * 8;   // B-tile copy coords (this thread)
  unsigned as_dst = to_lds(As + ra * 32 + ka);
  v8f acc[2] = {{}, {}};

  for (int ks = 0; ks < K; ks += 32) {
    // async copy A tile: one 16B global->LDS transfer per lane (ASYNCcnt)
    {
      const _Float16* asrc = A + (size_t)(m0 + ra) * K + ks + ka;
      asm volatile("global_load_async_to_lds_b128 %0, %1, off"
                   :: "v"(as_dst), "v"(asrc)
                   : "memory");
    }
    // B tile through registers (transpose in flight)
    {
      v8h gv = *(const v8h*)(B + (size_t)(ks + rb) * N + n0 + cb);
#pragma unroll
      for (int j = 0; j < 8; ++j) Bst[(cb + j) * 32 + rb] = gv[j];
    }
    // prefetch next K-step tiles while this step's WMMAs run
    if (ks + 32 < K) {
      __builtin_prefetch(A + (size_t)(m0 + ra) * K + ks + 32 + ka, 0, 3);
      __builtin_prefetch(B + (size_t)(ks + 32 + rb) * N + n0 + cb, 0, 3);
    }
    asm volatile("s_wait_asynccnt 0" ::: "memory");
    __syncthreads();

    v16h a = frag_a(As, 32, mt * 16, 0, lane);
#pragma unroll
    for (int j = 0; j < 2; ++j) {
      v16h bf = frag_bt(Bst, 32, 0, (ntb + j) * 16, lane);
      acc[j] = wmma_f16(a, bf, acc[j]);
    }
    __syncthreads();
  }

#pragma unroll
  for (int j = 0; j < 2; ++j) {
    int n     = n0 + (ntb + j) * 16 + (lane & 15);
    int mrow0 = m0 + mt * 16 + ((lane & 16) ? 8 : 0);
#pragma unroll
    for (int i = 0; i < 8; ++i) {
      int m = mrow0 + i;
      if (m < M) {
        float vl = acc[j][i] + bias[m];
        if (mode == 0) {
          out16[(size_t)m * N + n] =
              (_Float16)(0.5f * vl * (1.f + erff(vl * 0.70710678118f)));
        } else {
          int bb = n >> 14, p = n & (HWPIX - 1);
          size_t di = ((size_t)bb * CHN + m) * HWPIX + p;
          out32[di] = resid[di] + vl;
        }
      }
    }
  }
}

// -------------------------------- launcher ---------------------------------

extern "C" void kernel_launch(void* const* d_in, const int* in_sizes, int n_in,
                              void* d_out, int out_size, void* d_ws, size_t ws_size,
                              hipStream_t stream) {
  (void)in_sizes; (void)n_in; (void)out_size; (void)ws_size;
  const float* low     = (const float*)d_in[0];
  const float* high    = (const float*)d_in[1];
  const float* ln1_g   = (const float*)d_in[2];
  const float* ln1_b   = (const float*)d_in[3];
  const float* ln2_g   = (const float*)d_in[4];
  const float* ln2_b   = (const float*)d_in[5];
  const float* se_w1   = (const float*)d_in[6];
  const float* se_w2   = (const float*)d_in[7];
  const float* projw[6] = {(const float*)d_in[8],  (const float*)d_in[9],
                           (const float*)d_in[10], (const float*)d_in[11],
                           (const float*)d_in[12], (const float*)d_in[13]};
  const float* wproj_l = (const float*)d_in[14];
  const float* wproj_h = (const float*)d_in[15];
  const float* mlp_w1  = (const float*)d_in[16];
  const float* mlp_b1  = (const float*)d_in[17];
  const float* mlp_w2  = (const float*)d_in[18];
  const float* mlp_b2  = (const float*)d_in[19];
  float* dout = (float*)d_out;

  const size_t NE = (size_t)BATCH * CHN * HWPIX;   // 3145728
  char* ws = (char*)d_ws;
  size_t off = 0;
  auto take = [&](size_t bytes) -> char* {
    off = (off + 255) & ~(size_t)255;
    char* p = ws + off;
    off += bytes;
    return p;
  };
  float*    accL  = (float*)take(NE * 4);
  float*    accH  = (float*)take(NE * 4);
  _Float16* lnl   = (_Float16*)take(NE * 2);
  _Float16* lnh   = (_Float16*)take(NE * 2);
  float*    foldL = (float*)take(NE * 4);
  float*    foldH = (float*)take(NE * 4);
  float*    pool  = (float*)take(BATCH * CHN * 4);
  float*    sebuf = (float*)take(BATCH * CHN * 4);
  _Float16* wf16  = (_Float16*)take(6 * 9216 * 2);
  _Float16* w1f   = (_Float16*)take((size_t)384 * 96 * 2);
  _Float16* w2f   = (_Float16*)take((size_t)128 * 384 * 2);   // 96 rows + zero pad
  _Float16* X     = (_Float16*)take((size_t)CHN * BATCH * HWPIX * 2);
  _Float16* H1    = (_Float16*)take((size_t)384 * BATCH * HWPIX * 2);

  const int NPIX = BATCH * HWPIX;                 // 32768 pixels
  dim3 b256(256);

  copy_f32_kernel<<<dim3((unsigned)((NE + 255) / 256)), b256, 0, stream>>>(accL, low, NE);
  copy_f32_kernel<<<dim3((unsigned)((NE + 255) / 256)), b256, 0, stream>>>(accH, high, NE);

  for (int i = 0; i < 6; ++i)
    cvt_f16_kernel<<<dim3(36), b256, 0, stream>>>(wf16 + (size_t)i * 9216, projw[i], 9216);
  cvt_f16_kernel<<<dim3(144), b256, 0, stream>>>(w1f, mlp_w1, (size_t)384 * 96);
  zero_f16_kernel<<<dim3(192), b256, 0, stream>>>(w2f, (size_t)128 * 384);
  cvt_f16_kernel<<<dim3(144), b256, 0, stream>>>(w2f, mlp_w2, (size_t)96 * 384);

  const size_t attn_lds = (size_t)2 * 64 * CHN * 2 /*X low/high*/ +
                          (size_t)3 * 8192 * 2     /*qkv*/ +
                          (size_t)64 * 64 * 4      /*S*/ +
                          (size_t)64 * 64 * 2;     /*Sh*/   // = 98304 B

  for (int r = 0; r < 2; ++r) {
    int shift = r ? 4 : 0;
    const float* g  = r ? ln2_g : ln1_g;
    const float* bb = r ? ln2_b : ln1_b;
    ln_kernel<<<dim3(NPIX / 256), b256, 0, stream>>>(accL, g, bb, lnl);
    ln_kernel<<<dim3(NPIX / 256), b256, 0, stream>>>(accH, g, bb, lnh);
    pool_kernel<<<dim3(BATCH * CHN), b256, 0, stream>>>(lnh, pool);
    se_kernel<<<dim3(1), dim3(BATCH * CHN), 0, stream>>>(pool, se_w1, se_w2, sebuf);
    zero_f32_kernel<<<dim3((unsigned)((NE + 255) / 256)), b256, 0, stream>>>(foldL, NE);
    zero_f32_kernel<<<dim3((unsigned)((NE + 255) / 256)), b256, 0, stream>>>(foldH, NE);
    attn_kernel<<<dim3(BATCH * NWIN * NWIN), b256, attn_lds, stream>>>(
        lnl, lnh, sebuf, wf16, foldL, foldH, shift);
    proj_res_kernel<<<dim3(NPIX, 2), dim3(CHN), 0, stream>>>(
        foldL, foldH, wproj_l, wproj_h, lnl, lnh, accL, accH, shift);
  }

  for (int band = 0; band < 2; ++band) {
    const float* acc = band ? accH : accL;
    float* outp = dout + (size_t)band * NE;
    cvtX_kernel<<<dim3((unsigned)((NE + 255) / 256)), b256, 0, stream>>>(acc, X);
    gemm_kernel<<<dim3(6, NPIX / 64), b256, 0, stream>>>(
        w1f, X, mlp_b1, 384, NPIX, 96, 0, H1, nullptr, nullptr);
    gemm_kernel<<<dim3(2, NPIX / 64), b256, 0, stream>>>(
        w2f, H1, mlp_b2, 96, NPIX, 384, 1, nullptr, acc, outp);
  }
}
